// Diffusion_75797582840072
// MI455X (gfx1250) — matrix operand order repaired and checked by reading, not compiled, not run on hardware
//
#include <hip/hip_runtime.h>
#include <hip/hip_bf16.h>

typedef __attribute__((ext_vector_type(16))) _Float16 v16h;
typedef __attribute__((ext_vector_type(8)))  _Float16 v8h;
typedef __attribute__((ext_vector_type(8)))  float    v8f;
typedef __attribute__((ext_vector_type(4)))  unsigned int v4u;
typedef __attribute__((ext_vector_type(8)))  int      v8i_t;
typedef __attribute__((ext_vector_type(4)))  int      v4i_t;

union F16Frag { v16h v; v8h h[2]; };

// Fast mish: with u=e^x, tanh(softplus(x)) = (u^2+2u)/(u^2+2u+2).
__device__ __forceinline__ float mishf(float x) {
    float u = __expf(fminf(x, 30.f));
    float v = __builtin_fmaf(u, u, 2.f * u);
    float r = x * v * __builtin_amdgcn_rcpf(v + 2.f);
    return (x > 30.f) ? x : r;
}

__device__ __forceinline__ v8f wmma_f16(v16h a, v16h b, v8f c) {
    return __builtin_amdgcn_wmma_f32_16x16x32_f16(
        false, a, false, b, (short)0, c, false, false);
}

// LDS byte offset of a __shared__ object (AS3 ptrtoint)
__device__ __forceinline__ unsigned lds_addr_of(void* p) {
    typedef __attribute__((address_space(3))) void lds_void_t;
    return (unsigned)(unsigned long long)(lds_void_t*)p;
}

// Issue a TDM 2D tile load (tile_d0 x tile_d1 elements of 2 bytes) into LDS.
// Descriptor packing per CDNA5 ISA §8.3/8.4 (D# group0 + group1, groups 2/3 zero).
__device__ __forceinline__ void tdm_load_2d_f16(
    unsigned lds_byte_off, const _Float16* gptr,
    unsigned tensor_d0, unsigned tensor_d1,
    unsigned tile_d0, unsigned tile_d1, unsigned long long stride0_elems)
{
    unsigned long long ga = (unsigned long long)(uintptr_t)gptr;
    v4u g0;
    g0[0] = 1u;                                               // count=1, user mode
    g0[1] = lds_byte_off;                                     // lds_addr
    g0[2] = (unsigned)ga;                                     // global_addr[31:0]
    g0[3] = (unsigned)((ga >> 32) & 0x1FFFFFFull) | (2u << 30); // ga[56:32] | type=2
    v8i_t g1;
    g1[0] = (int)(1u << 16);                                  // data_size=1 (2 bytes)
    g1[1] = (int)((tensor_d0 & 0xFFFFu) << 16);               // tensor_dim0[15:0]
    g1[2] = (int)((tensor_d0 >> 16) | ((tensor_d1 & 0xFFFFu) << 16));
    g1[3] = (int)((tensor_d1 >> 16) | (tile_d0 << 16));       // tensor_dim1 hi | tile_dim0
    g1[4] = (int)(tile_d1 & 0xFFFFu);                         // tile_dim1, tile_dim2=0
    g1[5] = (int)(unsigned)(stride0_elems & 0xFFFFFFFFull);   // tensor_dim0_stride lo
    g1[6] = (int)(unsigned)((stride0_elems >> 32) & 0xFFFFull); // stride hi, stride1=0
    g1[7] = 0;
    v4i_t z4 = {0, 0, 0, 0};
    v8i_t z8 = {0, 0, 0, 0, 0, 0, 0, 0};
    __builtin_amdgcn_tensor_load_to_lds(g0, g1, z4, z4, z8, 0);
}

// ---------------------------------------------------------------------------
// Setup kernels
// ---------------------------------------------------------------------------

__global__ void sched_kernel(float* __restrict__ sched) {
    if (threadIdx.x != 0 || blockIdx.x != 0) return;
    float ab = 1.f;
    for (int i = 0; i < 100; ++i) {
        float beta  = 1e-4f + (0.2f - 1e-4f) * (float)i / 99.f;
        float alpha = 1.f - beta;
        float ab_prev = (i == 0) ? 1.f : ab;
        ab *= alpha;
        float post_var = beta * (1.f - ab_prev) / (1.f - ab);
        sched[0 * 100 + i] = sqrtf(1.f / ab);
        sched[1 * 100 + i] = sqrtf(1.f / ab - 1.f);
        sched[2 * 100 + i] = beta * sqrtf(ab_prev) / (1.f - ab);
        sched[3 * 100 + i] = (1.f - ab_prev) * sqrtf(alpha) / (1.f - ab);
        sched[4 * 100 + i] = sqrtf(fmaxf(post_var, 1e-20f));
    }
}

__global__ __launch_bounds__(256) void temb_kernel(
    const float* __restrict__ w_t1, const float* __restrict__ b_t1,
    const float* __restrict__ w_t2, const float* __restrict__ b_t2,
    _Float16* __restrict__ temb16)
{
    __shared__ float raw[16];
    __shared__ float hid[256];
    int i = blockIdx.x, tid = threadIdx.x;
    if (tid < 8) {
        float freq = __expf((float)tid * (-logf(10000.f) / 7.f));
        float ang = (float)i * freq;
        raw[tid]     = sinf(ang);
        raw[tid + 8] = cosf(ang);
    }
    __syncthreads();
    float a = b_t1[tid];
    #pragma unroll
    for (int j = 0; j < 16; ++j) a += raw[j] * w_t1[j * 256 + tid];
    hid[tid] = mishf(a);
    __syncthreads();
    if (tid < 16) {
        float o = b_t2[tid];
        for (int h = 0; h < 256; ++h) o += hid[h] * w_t2[h * 16 + tid];
        temb16[i * 16 + tid] = (_Float16)o;
    }
}

__global__ void cvt_transpose(const float* __restrict__ src, _Float16* __restrict__ dst,
                              int Krows, int Ncols, int Kpad)
{
    int idx = blockIdx.x * blockDim.x + threadIdx.x;
    if (idx >= Ncols * Kpad) return;
    int n = idx / Kpad, k = idx % Kpad;
    dst[(size_t)n * Kpad + k] = (k < Krows) ? (_Float16)src[(size_t)k * Ncols + n] : (_Float16)0.f;
}

__global__ void init_concat(const float* __restrict__ state, _Float16* __restrict__ hin)
{
    int idx = blockIdx.x * blockDim.x + threadIdx.x;
    if (idx >= 4096 * 272) return;
    int b = idx / 272, cc = idx % 272;
    hin[(size_t)b * 320 + 48 + cc] = (cc < 256) ? (_Float16)state[(size_t)b * 256 + cc]
                                               : (_Float16)0.f;
}

__global__ void step_prep(const float* __restrict__ x, const _Float16* __restrict__ temb16,
                          int t, _Float16* __restrict__ hin)
{
    int idx = blockIdx.x * blockDim.x + threadIdx.x;
    if (idx >= 4096 * 48) return;
    int b = idx / 48, c = idx % 48;
    _Float16 v = (c < 32) ? (_Float16)x[(size_t)b * 32 + c] : temb16[t * 16 + (c - 32)];
    hin[(size_t)b * 320 + c] = v;
}

// ---------------------------------------------------------------------------
// WMMA GEMM + mish with TDM tile staging.
// Out[M,N](f16) = mish( A[M,K](f16) @ Bt[N,K]^T + bias[N] )
// Block = 256 threads (8 waves), tile 128x128, stage K=64, double-buffered LDS.
// Wave 0 issues TDM loads for the next stage, then computes, then waits
// TENSORcnt; one workgroup barrier per stage.
// ---------------------------------------------------------------------------
__global__ __launch_bounds__(256) void gemm_mish_tdm(
    const _Float16* __restrict__ A, const _Float16* __restrict__ Bt,
    const float* __restrict__ bias, _Float16* __restrict__ Out,
    int M, int N, int K)
{
    __shared__ __attribute__((aligned(64))) _Float16 As[2][128 * 64];
    __shared__ __attribute__((aligned(64))) _Float16 Bs[2][128 * 64];

    int tid  = threadIdx.x;
    int lane = tid & 31;
    int wid  = tid >> 5;
    int m0 = blockIdx.x * 128;
    int n0 = blockIdx.y * 128;
    int wm = wid & 1;        // 0..1 -> rows [wm*64, +64)
    int wn = wid >> 1;       // 0..3 -> cols [wn*32, +32)

    v8f acc[4][2] = {};

    int r15  = lane & 15;
    int aoff = (lane >> 4) * 16;  // A frag: 8-half chunks at aoff, 8+aoff
    int boff = (lane >> 4) * 16;  // B frag: one contiguous 16-half chunk

    int nst = K >> 6;             // stages of K=64

    if (wid == 0) {
        tdm_load_2d_f16(lds_addr_of(&As[0][0]), A + (size_t)m0 * K, K, M, 64, 128, K);
        tdm_load_2d_f16(lds_addr_of(&Bs[0][0]), Bt + (size_t)n0 * K, K, N, 64, 128, K);
        __builtin_amdgcn_s_wait_tensorcnt((short)0);
    }
    __syncthreads();

    for (int ks = 0; ks < nst; ++ks) {
        int buf = ks & 1;
        if (wid == 0 && ks + 1 < nst) {
            tdm_load_2d_f16(lds_addr_of(&As[buf ^ 1][0]),
                            A + (size_t)m0 * K + (ks + 1) * 64, K, M, 64, 128, K);
            tdm_load_2d_f16(lds_addr_of(&Bs[buf ^ 1][0]),
                            Bt + (size_t)n0 * K + (ks + 1) * 64, K, N, 64, 128, K);
        }

        #pragma unroll
        for (int kk = 0; kk < 2; ++kk) {
            F16Frag af[4], bfr[2];
            #pragma unroll
            for (int i = 0; i < 4; ++i) {
                int row = wm * 64 + i * 16 + r15;
                af[i].h[0] = *(const v8h*)&As[buf][row * 64 + kk * 32 + aoff];
                af[i].h[1] = *(const v8h*)&As[buf][row * 64 + kk * 32 + 8 + aoff];
            }
            #pragma unroll
            for (int j = 0; j < 2; ++j) {
                int row = wn * 32 + j * 16 + r15;
                bfr[j].v = *(const v16h*)&Bs[buf][row * 64 + kk * 32 + boff];
            }
            #pragma unroll
            for (int i = 0; i < 4; ++i)
                #pragma unroll
                for (int j = 0; j < 2; ++j)
                    acc[i][j] = wmma_f16(af[i].v, bfr[j].v, acc[i][j]);
        }

        if (wid == 0 && ks + 1 < nst)
            __builtin_amdgcn_s_wait_tensorcnt((short)0);
        __syncthreads();
    }

    #pragma unroll
    for (int i = 0; i < 4; ++i) {
        #pragma unroll
        for (int j = 0; j < 2; ++j) {
            int n = n0 + wn * 32 + j * 16 + r15;
            float bv = bias[n];
            #pragma unroll
            for (int r = 0; r < 8; ++r) {
                int m = m0 + wm * 64 + i * 16 + r + 8 * (lane >> 4);
                Out[(size_t)m * N + n] = (_Float16)mishf(acc[i][j][r] + bv);
            }
        }
    }
}

// ---------------------------------------------------------------------------
// Final layer (N=32) fused with diffusion x-update (direct L2 loads).
// ---------------------------------------------------------------------------
__global__ __launch_bounds__(256) void final_step_kernel(
    const _Float16* __restrict__ A,    // h3  [4096 x 1024]
    const _Float16* __restrict__ Bt,   // wft [32 x 1024]
    const float* __restrict__ bf,
    float* __restrict__ x,             // [4096 x 32]
    const float* __restrict__ noise,   // [100 x 4096 x 32]
    const float* __restrict__ sched,   // [5 x 100]
    int s, int t)
{
    int tid  = threadIdx.x;
    int lane = tid & 31;
    int wid  = tid >> 5;
    int m0   = blockIdx.x * 128 + wid * 16;

    int r15  = lane & 15;
    int aoff = (lane >> 4) * 16;
    int boff = (lane >> 4) * 16;

    v8f acc[2] = {};
    for (int kb = 0; kb < 1024; kb += 32) {
        F16Frag a, b0, b1;
        const _Float16* ap = &A[(size_t)(m0 + r15) * 1024 + kb];
        a.h[0] = *(const v8h*)(ap + aoff);
        a.h[1] = *(const v8h*)(ap + 8 + aoff);
        b0.v = *(const v16h*)&Bt[(size_t)r15 * 1024 + kb + boff];
        b1.v = *(const v16h*)&Bt[(size_t)(16 + r15) * 1024 + kb + boff];
        acc[0] = wmma_f16(a.v, b0.v, acc[0]);
        acc[1] = wmma_f16(a.v, b1.v, acc[1]);
    }

    float srec  = sched[0 * 100 + t];
    float srecm = sched[1 * 100 + t];
    float p1    = sched[2 * 100 + t];
    float p2    = sched[3 * 100 + t];
    float sig   = (t != 0) ? sched[4 * 100 + t] : 0.f;
    const float* z = &noise[(size_t)s * 4096 * 32];

    #pragma unroll
    for (int j = 0; j < 2; ++j) {
        #pragma unroll
        for (int r = 0; r < 8; ++r) {
            int n = j * 16 + r15;
            int m = m0 + r + 8 * (lane >> 4);
            size_t idx = (size_t)m * 32 + n;
            float eps = acc[j][r] + bf[n];
            float xo = x[idx];
            float xr = fminf(fmaxf(srec * xo - srecm * eps, -1.f), 1.f);
            x[idx] = p1 * xr + p2 * xo + sig * z[idx];
        }
    }
}

__global__ void out_clip(const float* __restrict__ x, float* __restrict__ out)
{
    int idx = blockIdx.x * blockDim.x + threadIdx.x;
    if (idx >= 4096 * 32) return;
    out[idx] = fminf(fmaxf(x[idx], -1.f), 1.f);
}

// ---------------------------------------------------------------------------
extern "C" void kernel_launch(void* const* d_in, const int* in_sizes, int n_in,
                              void* d_out, int out_size, void* d_ws, size_t ws_size,
                              hipStream_t stream) {
    (void)in_sizes; (void)n_in; (void)out_size; (void)ws_size;

    const float* state  = (const float*)d_in[0];
    const float* w_t1   = (const float*)d_in[1];
    const float* b_t1   = (const float*)d_in[2];
    const float* w_t2   = (const float*)d_in[3];
    const float* b_t2   = (const float*)d_in[4];
    const float* w1     = (const float*)d_in[5];
    const float* b1     = (const float*)d_in[6];
    const float* w2     = (const float*)d_in[7];
    const float* b2     = (const float*)d_in[8];
    const float* w3     = (const float*)d_in[9];
    const float* b3     = (const float*)d_in[10];
    const float* wf     = (const float*)d_in[11];
    const float* bfv    = (const float*)d_in[12];
    const float* x_init = (const float*)d_in[13];
    const float* noise  = (const float*)d_in[14];

    char* wsb = (char*)d_ws;
    size_t off = 0;
    auto alloc = [&](size_t bytes) -> void* {
        void* p = wsb + off;
        off = (off + bytes + 255) & ~(size_t)255;
        return p;
    };

    _Float16* w1t    = (_Float16*)alloc((size_t)1024 * 320 * 2);
    _Float16* w2t    = (_Float16*)alloc((size_t)1024 * 1024 * 2);
    _Float16* w3t    = (_Float16*)alloc((size_t)1024 * 1024 * 2);
    _Float16* wft    = (_Float16*)alloc((size_t)32 * 1024 * 2);
    _Float16* temb16 = (_Float16*)alloc((size_t)100 * 16 * 2);
    float*    sched  = (float*)alloc((size_t)5 * 100 * 4);
    _Float16* hin    = (_Float16*)alloc((size_t)4096 * 320 * 2);
    _Float16* hA     = (_Float16*)alloc((size_t)4096 * 1024 * 2);
    _Float16* hB     = (_Float16*)alloc((size_t)4096 * 1024 * 2);
    float*    xbuf   = (float*)alloc((size_t)4096 * 32 * 4);

    sched_kernel<<<1, 32, 0, stream>>>(sched);
    temb_kernel<<<100, 256, 0, stream>>>(w_t1, b_t1, w_t2, b_t2, temb16);
    cvt_transpose<<<(1024 * 320 + 255) / 256, 256, 0, stream>>>(w1, w1t, 304, 1024, 320);
    cvt_transpose<<<(1024 * 1024 + 255) / 256, 256, 0, stream>>>(w2, w2t, 1024, 1024, 1024);
    cvt_transpose<<<(1024 * 1024 + 255) / 256, 256, 0, stream>>>(w3, w3t, 1024, 1024, 1024);
    cvt_transpose<<<(32 * 1024 + 255) / 256, 256, 0, stream>>>(wf, wft, 1024, 32, 1024);
    init_concat<<<(4096 * 272 + 255) / 256, 256, 0, stream>>>(state, hin);
    (void)hipMemcpyAsync(xbuf, x_init, (size_t)4096 * 32 * 4,
                         hipMemcpyDeviceToDevice, stream);

    for (int s = 0; s < 100; ++s) {
        int t = 99 - s;
        step_prep<<<(4096 * 48 + 255) / 256, 256, 0, stream>>>(xbuf, temb16, t, hin);
        gemm_mish_tdm<<<dim3(32, 8), 256, 0, stream>>>(hin, w1t, b1, hA, 4096, 1024, 320);
        gemm_mish_tdm<<<dim3(32, 8), 256, 0, stream>>>(hA, w2t, b2, hB, 4096, 1024, 1024);
        gemm_mish_tdm<<<dim3(32, 8), 256, 0, stream>>>(hB, w3t, b3, hA, 4096, 1024, 1024);
        final_step_kernel<<<32, 256, 0, stream>>>(hA, wft, bfv, xbuf, noise, sched, s, t);
    }

    out_clip<<<(4096 * 32 + 255) / 256, 256, 0, stream>>>(xbuf, (float*)d_out);
}
